// GMambaBlock_63178968924529
// MI455X (gfx1250) — matrix operand order repaired and checked by reading, not compiled, not run on hardware
//
#include <hip/hip_runtime.h>
#include <hip/hip_bf16.h>
#include <stdint.h>

// ---------------------------------------------------------------------------
// Problem constants (from the reference)
// ---------------------------------------------------------------------------
#define Hdim   512
#define SEQ    4096
#define NBATCH 4
#define TTOK   (NBATCH * SEQ)       // 16384 tokens
#define INNER  1024
#define NFLIP  (SEQ / 2)            // partial_flip reverses first 2048 positions

typedef unsigned short u16;         // bf16 storage
typedef __attribute__((ext_vector_type(16))) __bf16 v16bf;
typedef __attribute__((ext_vector_type(8)))  float  v8f;
typedef __attribute__((ext_vector_type(4)))  unsigned int v4u;
typedef __attribute__((ext_vector_type(8)))  int v8i;
typedef __attribute__((ext_vector_type(4)))  int v4i;

#define AS1 __attribute__((address_space(1)))
#define AS3 __attribute__((address_space(3)))

// Feature probes (device pass only; fallbacks keep host pass + old toolchains OK)
#if defined(__HIP_DEVICE_COMPILE__) && __has_builtin(__builtin_amdgcn_global_load_async_to_lds_b128)
#define HAVE_ASYNC_LDS 1
#else
#define HAVE_ASYNC_LDS 0
#endif
#if defined(__HIP_DEVICE_COMPILE__) && __has_builtin(__builtin_amdgcn_tensor_load_to_lds) && \
    __has_builtin(__builtin_amdgcn_s_wait_tensorcnt)
#define HAVE_TDM 1
#else
#define HAVE_TDM 0
#endif
#if defined(__HIP_DEVICE_COMPILE__) && __has_builtin(__builtin_amdgcn_s_wait_asynccnt)
#define HAVE_WAIT_ASYNC 1
#else
#define HAVE_WAIT_ASYNC 0
#endif

// ---------------------------------------------------------------------------
// Small device helpers
// ---------------------------------------------------------------------------
__device__ __forceinline__ u16 f2bf(float f) {
  union { float f; uint32_t u; } v; v.f = f;
  uint32_t u = v.u + 0x7FFFu + ((v.u >> 16) & 1u);   // round-to-nearest-even
  return (u16)(u >> 16);
}
__device__ __forceinline__ __bf16 u2bf(u16 x) {
  union { u16 u; __bf16 h; } c; c.u = x; return c.h;
}
__device__ __forceinline__ float sigm(float x) { return 1.f / (1.f + __expf(-x)); }

// ---------------------------------------------------------------------------
// Generic bf16 WMMA GEMM:  C[m,n] (+)= act( sum_k A[m,k] * W[n,k] + bias[n] )
//   A: bf16 row-major (ldA); rows can be shifted by `shift` inside each batch
//      sequence (zero outside) -> "same" convs become 3 accumulated GEMMs.
//   W: bf16 row-major (N x K = weight (out,in)), leading dim ldB.
//   C: fp32 (ldC); optional dual bf16 output Cbf; act: 0 none,1 sigmoid,2 silu
// Block = 256 threads = 8 waves (wave32). Block tile 128x64, K-step 32.
// Each wave computes a 32x32 tile = 2x2 v_wmma_f32_16x16x32_bf16.
// A tile staged via global_load_async_to_lds_b128 (ASYNCcnt);
// B tile staged via TDM tensor_load_to_lds (TENSORcnt) issued by wave 0.
// ---------------------------------------------------------------------------
#define BM 128
#define BN 64
#define BK 32

__global__ void __launch_bounds__(256)
gemm_bf16_wmma(const u16* __restrict__ A, int ldA,
               const u16* __restrict__ Bw, int ldB,
               float* __restrict__ C, int ldC,
               u16* __restrict__ Cbf,
               const float* __restrict__ bias,
               int M, int N, int K,
               int S, int shift, int accum, int act)
{
  __shared__ __align__(16) u16 As[BM * BK];
  __shared__ __align__(16) u16 Bs[BN * BK];

  const int tid    = threadIdx.x;
  const int wave   = tid >> 5;
  const int lane   = tid & 31;
  const int wm     = wave >> 1;        // 0..3 (M direction)
  const int wn     = wave & 1;         // 0..1 (N direction)
  const int laneM  = lane & 15;
  const int laneHi = lane >> 4;
  const int rowBase = blockIdx.y * BM;
  const int colBase = blockIdx.x * BN;

  v8f acc[2][2];
  const v8f vzero = {0.f,0.f,0.f,0.f,0.f,0.f,0.f,0.f};
  acc[0][0] = vzero; acc[0][1] = vzero; acc[1][0] = vzero; acc[1][1] = vzero;

  for (int k0 = 0; k0 < K; k0 += BK) {
    // ---- stage A tile (128x32 bf16): 512 chunks of 16B; 2 chunks per thread
#if HAVE_ASYNC_LDS
    #pragma unroll
    for (int j = 0; j < 2; ++j) {
      int ch = tid + j * 256;
      int r  = ch >> 2;                 // 0..127
      int kc = (ch & 3) * 8;
      int t  = rowBase + r;
      int b  = t / S;
      int s  = (t % S) + shift;
      if (s >= 0 && s < S) {
        const u16* src = A + ((size_t)b * S + s) * (size_t)ldA + k0 + kc;
        __builtin_amdgcn_global_load_async_to_lds_b128(
            (AS1 v4i*)(uintptr_t)src,
            (AS3 v4i*)(uint32_t)(uintptr_t)&As[r * BK + kc], 0, 0);
      } else {
        *(uint4*)(&As[r * BK + kc]) = make_uint4(0u, 0u, 0u, 0u);
      }
    }
#else
    #pragma unroll
    for (int j = 0; j < 2; ++j) {
      int ch = tid + j * 256;
      int r  = ch >> 2;
      int kc = (ch & 3) * 8;
      int t  = rowBase + r;
      int b  = t / S;
      int s  = (t % S) + shift;
      uint4 val = make_uint4(0u, 0u, 0u, 0u);
      if (s >= 0 && s < S) {
        const u16* src = A + ((size_t)b * S + s) * (size_t)ldA + k0 + kc;
        val = *(const uint4*)src;
        if (k0 + BK < K) __builtin_prefetch(src + BK, 0, 3);
      }
      *(uint4*)(&As[r * BK + kc]) = val;
    }
#endif

    // ---- stage B tile (64x32 bf16)
#if HAVE_TDM
    if (wave == 0) {
      // Tensor DMA descriptor (ISA 8.3/8.4): 2D tile, data_size=2B,
      // tile 32(K) x 64(N), tensor bounds clamp/zero OOB rows.
      unsigned long long ga =
          (unsigned long long)(uintptr_t)(Bw + (size_t)colBase * (size_t)ldB + k0);
      unsigned int laddr = (uint32_t)(uintptr_t)&Bs[0];
      int remK = K - k0;
      int remN = N - colBase;
      v4u g0;
      g0[0] = 1u;                                            // count=1 (valid D#)
      g0[1] = laddr;                                         // lds_addr
      g0[2] = (unsigned int)(ga & 0xFFFFFFFFull);            // global_addr[31:0]
      g0[3] = (unsigned int)((ga >> 32) & 0x01FFFFFFull)     // global_addr[56:32]
              | 0x80000000u;                                 // type=2 ("image")
      v8i g1;
      g1[0] = (int)(1u << 16);                               // data_size=1 -> 2 bytes
      g1[1] = (int)(((unsigned)remK & 0xFFFFu) << 16);       // tensor_dim0[15:0]
      g1[2] = (int)((((unsigned)remK >> 16) & 0xFFFFu)       // tensor_dim0[31:16]
              | (((unsigned)remN & 0xFFFFu) << 16));         // tensor_dim1[15:0]
      g1[3] = (int)((((unsigned)remN >> 16) & 0xFFFFu)       // tensor_dim1[31:16]
              | ((unsigned)BK << 16));                       // tile_dim0 = 32
      g1[4] = BN;                                            // tile_dim1=64, tile_dim2=0
      g1[5] = ldB;                                           // tensor_dim0_stride[31:0]
      g1[6] = 0;                                             // stride hi / dim1_stride lo
      g1[7] = 0;
      v4i gz = {0, 0, 0, 0};
#if __clang_major__ >= 23
      v8i gz8 = {0, 0, 0, 0, 0, 0, 0, 0};
      __builtin_amdgcn_tensor_load_to_lds(g0, g1, gz, gz, gz8, 0);
#else
      __builtin_amdgcn_tensor_load_to_lds(g0, g1, gz, gz, 0);
#endif
      __builtin_amdgcn_s_wait_tensorcnt(0);
    }
#else
    {
      int n  = tid >> 2;
      int kc = (tid & 3) * 8;
      int ng = colBase + n;
      uint4 val = make_uint4(0u, 0u, 0u, 0u);
      if (ng < N) {
        const u16* src = Bw + (size_t)ng * (size_t)ldB + k0 + kc;
        val = *(const uint4*)src;
        if (k0 + BK < K) __builtin_prefetch(src + BK, 0, 3);
      }
      *(uint4*)(&Bs[n * BK + kc]) = val;
    }
#endif

#if HAVE_ASYNC_LDS
#if HAVE_WAIT_ASYNC
    __builtin_amdgcn_s_wait_asynccnt(0);
#else
    asm volatile("s_wait_asynccnt 0x0" ::: "memory");
#endif
#endif
    __syncthreads();

    // ---- build fragments per ISA 7.12.2 wave32 layouts
    v16bf afrag[2], bfrag[2];
    #pragma unroll
    for (int mi = 0; mi < 2; ++mi) {
      int m = wm * 32 + mi * 16 + laneM;
      const u16* pr = &As[m * BK];
      #pragma unroll
      for (int e = 0; e < 8; ++e) {
        afrag[mi][e]     = u2bf(pr[laneHi * 8 + e]);        // K = hi*8 + e
        afrag[mi][e + 8] = u2bf(pr[16 + laneHi * 8 + e]);   // K = 16 + hi*8 + e
      }
    }
    #pragma unroll
    for (int ni = 0; ni < 2; ++ni) {
      int n = wn * 32 + ni * 16 + laneM;
      const u16* pr = &Bs[n * BK];
      #pragma unroll
      for (int e = 0; e < 16; ++e)
        bfrag[ni][e] = u2bf(pr[(e & 8) * 2 + laneHi * 8 + (e & 7)]);  // K = (e>=8)*16 + hi*8 + e%8
    }

    #pragma unroll
    for (int mi = 0; mi < 2; ++mi)
      #pragma unroll
      for (int ni = 0; ni < 2; ++ni)
        acc[mi][ni] = __builtin_amdgcn_wmma_f32_16x16x32_bf16(
            false, afrag[mi], false, bfrag[ni], (short)0, acc[mi][ni],
            false, false);
    __syncthreads();
  }

  // ---- epilogue (C/D layout: VGPR v holds M = v + laneHi*8, N = laneM)
  #pragma unroll
  for (int mi = 0; mi < 2; ++mi) {
    #pragma unroll
    for (int ni = 0; ni < 2; ++ni) {
      int cn = colBase + wn * 32 + ni * 16 + laneM;
      if (cn >= N) continue;
      #pragma unroll
      for (int v = 0; v < 8; ++v) {
        int ml = wm * 32 + mi * 16 + v + laneHi * 8;
        size_t idx = ((size_t)(rowBase + ml)) * (size_t)ldC + cn;
        float val = acc[mi][ni][v];
        if (accum) val += C[idx];
        if (bias)  val += bias[cn];
        if (act == 1)      val = sigm(val);
        else if (act == 2) val = val * sigm(val);
        C[idx] = val;
        if (Cbf) Cbf[idx] = f2bf(val);
      }
    }
  }
}

// ---------------------------------------------------------------------------
// Converters / reshapers
// ---------------------------------------------------------------------------
__global__ void cvt_gather_kernel(u16* dst, const float* src, int n, int stride, int off) {
  int i = blockIdx.x * blockDim.x + threadIdx.x;
  if (i < n) dst[i] = f2bf(src[(size_t)i * stride + off]);
}

__global__ void cvt2d_kernel(u16* dst, const float* src, int rows, int cols,
                             int srcld, int srcoff) {
  int gid = blockIdx.x * blockDim.x + threadIdx.x;
  if (gid >= rows * cols) return;
  int r = gid / cols, c = gid % cols;
  dst[gid] = f2bf(src[(size_t)r * srcld + srcoff + c]);
}

__global__ void flip_cvt_kernel(u16* dst, const float* src) {
  int gid = blockIdx.x * blockDim.x + threadIdx.x;
  if (gid >= TTOK * Hdim) return;
  int c = gid % Hdim, t = gid / Hdim;
  int b = t / SEQ, s = t % SEQ;
  int sf = (s < NFLIP) ? (NFLIP - 1 - s) : s;
  dst[gid] = f2bf(src[((size_t)b * SEQ + sf) * Hdim + c]);
}

// ---------------------------------------------------------------------------
// Depthwise causal conv (K=4) + SiLU.  Input = x_proj view of xz (ld 2048).
// ---------------------------------------------------------------------------
__global__ void dwconv_silu_kernel(const float* __restrict__ xz,
                                   const float* __restrict__ cw,
                                   const float* __restrict__ cb,
                                   float* __restrict__ xconv,
                                   u16* __restrict__ xconvbf)
{
  int gid = blockIdx.x * blockDim.x + threadIdx.x;
  if (gid >= TTOK * INNER) return;
  int c = gid & (INNER - 1);
  int t = gid >> 10;
  int b = t / SEQ, s = t % SEQ;
  float acc = cb[c];
  #pragma unroll
  for (int k = 0; k < 4; ++k) {
    int ss = s + k - 3;
    if (ss >= 0) acc += cw[c * 4 + k] * xz[((size_t)b * SEQ + ss) * 2048 + c];
  }
  float y = acc * sigm(acc);
  xconv[gid]   = y;
  xconvbf[gid] = f2bf(y);
}

// ---------------------------------------------------------------------------
// B_gate / C_gate: sigmoid of mean over the 16 B / C state columns of xs
// ---------------------------------------------------------------------------
__global__ void bcgate_kernel(const float* __restrict__ xs,
                              float* __restrict__ Bg, float* __restrict__ Cg) {
  int t = blockIdx.x * blockDim.x + threadIdx.x;
  if (t >= TTOK) return;
  const float* r = xs + (size_t)t * 1056;
  float sb = 0.f, sc = 0.f;
  #pragma unroll
  for (int i = 0; i < 16; ++i) { sb += r[i]; sc += r[16 + i]; }
  Bg[t] = sigm(sb * (1.f / 16.f));
  Cg[t] = sigm(sc * (1.f / 16.f));
}

// ---------------------------------------------------------------------------
// Recurrent scan: one thread per (batch, channel) chain, fused epilogue.
// ---------------------------------------------------------------------------
__global__ void scan_kernel(const float* __restrict__ dtraw,
                            const float* __restrict__ xconv,
                            const float* __restrict__ xz,  // z at col 1024, ld 2048
                            const float* __restrict__ Bg,
                            const float* __restrict__ Cg,
                            const float* __restrict__ Dp,
                            u16* __restrict__ ybf)
{
  int gid = blockIdx.x * blockDim.x + threadIdx.x;
  if (gid >= NBATCH * INNER) return;
  int b = gid >> 10;
  int c = gid & (INNER - 1);
  float h = 0.f;
  float Dv = Dp[c];
  for (int s = 0; s < SEQ; ++s) {
    size_t t   = (size_t)b * SEQ + s;
    size_t idx = t * INNER + c;
    if (s + 1 < SEQ) {
      __builtin_prefetch(&dtraw[idx + INNER], 0, 3);
      __builtin_prefetch(&xconv[idx + INNER], 0, 3);
    }
    float xr = dtraw[idx];
    float sp = (xr > 20.f) ? xr : log1pf(__expf(xr));
    float alpha = sigm(0.1f * sp);
    float ac = fminf(fmaxf(alpha, 0.01f), 0.99f);
    float xc = xconv[idx];
    h = ac * h + (1.f - alpha) * Bg[t] * xc;
    float z = xz[t * 2048 + INNER + c];
    float y = (h * Cg[t] + Dv * xc) * (z * sigm(z));
    ybf[idx] = f2bf(y);
  }
}

// ---------------------------------------------------------------------------
// m_comb = gw * m_fwd + (1-gw) * unflip(m_bwd)
// ---------------------------------------------------------------------------
__global__ void combine_mamba_kernel(const float* __restrict__ gw,
                                     const float* __restrict__ mf,
                                     const float* __restrict__ mb,
                                     float* __restrict__ mcomb,
                                     u16* __restrict__ mcombbf)
{
  int gid = blockIdx.x * blockDim.x + threadIdx.x;
  if (gid >= TTOK * Hdim) return;
  int c = gid % Hdim, t = gid / Hdim;
  int b = t / SEQ, s = t % SEQ;
  int sf = (s < NFLIP) ? (NFLIP - 1 - s) : s;
  float mbv = mb[((size_t)b * SEQ + sf) * Hdim + c];
  float g = gw[gid];
  float v = g * mf[gid] + (1.f - g) * mbv;
  mcomb[gid]   = v;
  mcombbf[gid] = f2bf(v);
}

// z = g*a + (1-g)*b
__global__ void combine2_kernel(const float* __restrict__ g,
                                const float* __restrict__ a,
                                const float* __restrict__ b,
                                float* __restrict__ out, u16* __restrict__ outbf, int n)
{
  int gid = blockIdx.x * blockDim.x + threadIdx.x;
  if (gid >= n) return;
  float gg = g[gid];
  float v = gg * a[gid] + (1.f - gg) * b[gid];
  out[gid] = v;
  if (outbf) outbf[gid] = f2bf(v);
}

// ---------------------------------------------------------------------------
// out = LayerNorm(A + X) * gamma + beta ;  one wave32 per row of 512
// ---------------------------------------------------------------------------
__global__ void __launch_bounds__(256)
layernorm_res_kernel(const float* __restrict__ A, const float* __restrict__ X,
                     const float* __restrict__ gam, const float* __restrict__ bet,
                     float* __restrict__ out, u16* __restrict__ outbf)
{
  int wave = threadIdx.x >> 5, lane = threadIdx.x & 31;
  int row = blockIdx.x * 8 + wave;
  if (row >= TTOK) return;
  const float* a = A + (size_t)row * Hdim;
  const float* x = X + (size_t)row * Hdim;
  float vals[16];
  float s = 0.f;
  #pragma unroll
  for (int i = 0; i < 16; ++i) {
    float v = a[lane + i * 32] + x[lane + i * 32];
    vals[i] = v; s += v;
  }
  #pragma unroll
  for (int off = 16; off > 0; off >>= 1) s += __shfl_xor(s, off, 32);
  float mean = s * (1.f / (float)Hdim);
  float vs = 0.f;
  #pragma unroll
  for (int i = 0; i < 16; ++i) { float d = vals[i] - mean; vs += d * d; }
  #pragma unroll
  for (int off = 16; off > 0; off >>= 1) vs += __shfl_xor(vs, off, 32);
  float inv = rsqrtf(vs * (1.f / (float)Hdim) + 1e-5f);
  #pragma unroll
  for (int i = 0; i < 16; ++i) {
    int c = lane + i * 32;
    float o = (vals[i] - mean) * inv * gam[c] + bet[c];
    out[(size_t)row * Hdim + c] = o;
    if (outbf) outbf[(size_t)row * Hdim + c] = f2bf(o);
  }
}

// ---------------------------------------------------------------------------
// Host orchestration
// ---------------------------------------------------------------------------
extern "C" void kernel_launch(void* const* d_in, const int* in_sizes, int n_in,
                              void* d_out, int out_size, void* d_ws, size_t ws_size,
                              hipStream_t stream)
{
  const float* x      = (const float*)d_in[0];
  const float* ds_w   = (const float*)d_in[2];
  const float* ds_b   = (const float*)d_in[3];
  const float* gate_w = (const float*)d_in[4];
  const float* gate_b = (const float*)d_in[5];
  const float* mix_w  = (const float*)d_in[6];
  const float* mix_b  = (const float*)d_in[7];
  const float* ln_g   = (const float*)d_in[8];
  const float* ln_b   = (const float*)d_in[9];
  const float* fe1_w  = (const float*)d_in[10];
  const float* fe1_b  = (const float*)d_in[11];
  const float* fe2_w  = (const float*)d_in[12];
  const float* fe2_b  = (const float*)d_in[13];
  const float* feln_g = (const float*)d_in[14];
  const float* feln_b = (const float*)d_in[15];
  const float* in_w[2]   = {(const float*)d_in[16], (const float*)d_in[24]};
  const float* conv_w[2] = {(const float*)d_in[17], (const float*)d_in[25]};
  const float* conv_b[2] = {(const float*)d_in[18], (const float*)d_in[26]};
  const float* xp_w[2]   = {(const float*)d_in[19], (const float*)d_in[27]};
  const float* dt_w[2]   = {(const float*)d_in[20], (const float*)d_in[28]};
  const float* dt_b[2]   = {(const float*)d_in[21], (const float*)d_in[29]};
  const float* Dp[2]     = {(const float*)d_in[22], (const float*)d_in[30]};
  const float* out_w[2]  = {(const float*)d_in[23], (const float*)d_in[31]};

  // bump allocator over d_ws (256B aligned carves)
  char* p = (char*)d_ws;
  auto carve = [&](size_t bytes) -> char* {
    char* r = p; p += (bytes + 255) & ~(size_t)255; return r;
  };
  const size_t T = TTOK;

  u16* xb      = (u16*)carve(T * Hdim * 2);
  u16* xfb     = (u16*)carve(T * Hdim * 2);
  u16* w_ds    = (u16*)carve((size_t)Hdim * Hdim * 2);
  u16* w_gate  = (u16*)carve((size_t)Hdim * 2 * Hdim * 2);
  u16* w_mix   = (u16*)carve((size_t)Hdim * Hdim * 2);
  u16* w_fe1[3], *w_fe2[3];
  for (int t = 0; t < 3; ++t) w_fe1[t] = (u16*)carve((size_t)Hdim * Hdim * 2);
  for (int t = 0; t < 3; ++t) w_fe2[t] = (u16*)carve((size_t)Hdim * Hdim * 2);
  u16 *w_in[2], *w_xp[2], *w_dt[2], *w_out[2];
  for (int d = 0; d < 2; ++d) {
    w_in[d]  = (u16*)carve((size_t)2 * INNER * Hdim * 2);
    w_xp[d]  = (u16*)carve((size_t)1056 * INNER * 2);
    w_dt[d]  = (u16*)carve((size_t)INNER * INNER * 2);
    w_out[d] = (u16*)carve((size_t)Hdim * INNER * 2);
  }
  float* gw      = (float*)carve(T * Hdim * 4);
  float* xz      = (float*)carve(T * 2 * INNER * 4);
  float* xconv   = (float*)carve(T * INNER * 4);
  u16*   xconvbf = (u16*)  carve(T * INNER * 2);
  float* xs      = (float*)carve(T * 1056 * 4);
  float* Bg      = (float*)carve(T * 4);
  float* Cg      = (float*)carve(T * 4);
  u16*   dtinbf  = (u16*)  carve(T * INNER * 2);
  float* dtraw   = (float*)carve(T * INNER * 4);
  u16*   ybf     = (u16*)  carve(T * INNER * 2);
  float* mf      = (float*)carve(T * Hdim * 4);
  float* mb      = (float*)carve(T * Hdim * 4);
  float* mcomb   = (float*)carve(T * Hdim * 4);
  u16*   mcombbf = (u16*)  carve(T * Hdim * 2);
  float* c1      = (float*)carve(T * Hdim * 4);
  u16*   c1bf    = (u16*)  carve(T * Hdim * 2);
  float* c2      = (float*)carve(T * Hdim * 4);
  float* fout    = (float*)carve(T * Hdim * 4);
  u16*   foutbf  = (u16*)  carve(T * Hdim * 2);
  float* gbuf    = (float*)carve(T * Hdim * 4);
  float* zsum    = (float*)carve(T * Hdim * 4);
  u16*   zsumbf  = (u16*)  carve(T * Hdim * 2);
  float* z2      = (float*)carve(T * Hdim * 4);

  auto cvt = [&](u16* dst, const float* src, int n, int stride, int off) {
    cvt_gather_kernel<<<(n + 255) / 256, 256, 0, stream>>>(dst, src, n, stride, off);
  };
  auto gemm = [&](const u16* A, int ldA, const u16* Bw, int ldB,
                  float* C, int ldC, u16* Cbf, const float* bias,
                  int N, int K, int shift, int accum, int act) {
    dim3 grid((N + BN - 1) / BN, TTOK / BM);
    gemm_bf16_wmma<<<grid, 256, 0, stream>>>(A, ldA, Bw, ldB, C, ldC, Cbf, bias,
                                             TTOK, N, K, SEQ, shift, accum, act);
  };

  // ---- stage 0: conversions
  cvt(xb, x, TTOK * Hdim, 1, 0);
  flip_cvt_kernel<<<(TTOK * Hdim + 255) / 256, 256, 0, stream>>>(xfb, x);
  cvt(w_ds,   ds_w,   Hdim * Hdim,     1, 0);
  cvt(w_gate, gate_w, Hdim * 2 * Hdim, 1, 0);
  cvt(w_mix,  mix_w,  Hdim * Hdim,     1, 0);
  for (int t = 0; t < 3; ++t) cvt(w_fe1[t], fe1_w, Hdim * Hdim, 3, t);   // de-interleave OIH
  for (int t = 0; t < 3; ++t) cvt(w_fe2[t], fe2_w, Hdim * Hdim, 3, t);
  for (int d = 0; d < 2; ++d) {
    cvt(w_in[d],  in_w[d],  2 * INNER * Hdim, 1, 0);
    cvt(w_xp[d],  xp_w[d],  1056 * INNER,     1, 0);
    cvt(w_dt[d],  dt_w[d],  INNER * INNER,    1, 0);
    cvt(w_out[d], out_w[d], Hdim * INNER,     1, 0);
  }

  // ---- stage 1: gate_weights = sigmoid(x @ ds_w^T + ds_b)
  gemm(xb, Hdim, w_ds, Hdim, gw, Hdim, nullptr, ds_b, Hdim, Hdim, 0, 0, 1);

  // ---- feature-extractor branch: two "same" convs as 3 shifted GEMMs each
  gemm(xb, Hdim, w_fe1[0], Hdim, c1, Hdim, nullptr, fe1_b, Hdim, Hdim, -1, 0, 0);
  gemm(xb, Hdim, w_fe1[1], Hdim, c1, Hdim, nullptr, nullptr, Hdim, Hdim, 0, 1, 0);
  gemm(xb, Hdim, w_fe1[2], Hdim, c1, Hdim, c1bf,    nullptr, Hdim, Hdim, 1, 1, 2); // silu
  gemm(c1bf, Hdim, w_fe2[0], Hdim, c2, Hdim, nullptr, fe2_b, Hdim, Hdim, -1, 0, 0);
  gemm(c1bf, Hdim, w_fe2[1], Hdim, c2, Hdim, nullptr, nullptr, Hdim, Hdim, 0, 1, 0);
  gemm(c1bf, Hdim, w_fe2[2], Hdim, c2, Hdim, nullptr, nullptr, Hdim, Hdim, 1, 1, 0);
  layernorm_res_kernel<<<TTOK / 8, 256, 0, stream>>>(c2, x, feln_g, feln_b, fout, foutbf);

  // ---- mamba blocks (fwd d=0 on x, bwd d=1 on partial_flip(x))
  for (int d = 0; d < 2; ++d) {
    const u16* xin = d ? xfb : xb;
    gemm(xin, Hdim, w_in[d], Hdim, xz, 2 * INNER, nullptr, nullptr, 2 * INNER, Hdim, 0, 0, 0);
    dwconv_silu_kernel<<<(TTOK * INNER + 255) / 256, 256, 0, stream>>>(
        xz, conv_w[d], conv_b[d], xconv, xconvbf);
    gemm(xconvbf, INNER, w_xp[d], INNER, xs, 1056, nullptr, nullptr, 1056, INNER, 0, 0, 0);
    bcgate_kernel<<<(TTOK + 255) / 256, 256, 0, stream>>>(xs, Bg, Cg);
    cvt2d_kernel<<<(TTOK * INNER + 255) / 256, 256, 0, stream>>>(
        dtinbf, xs, TTOK, INNER, 1056, 32);
    gemm(dtinbf, INNER, w_dt[d], INNER, dtraw, INNER, nullptr, dt_b[d], INNER, INNER, 0, 0, 0);
    scan_kernel<<<(NBATCH * INNER) / 256, 256, 0, stream>>>(
        dtraw, xconv, xz, Bg, Cg, Dp[d], ybf);
    gemm(ybf, INNER, w_out[d], INNER, d ? mb : mf, Hdim, nullptr, nullptr, Hdim, INNER, 0, 0, 0);
  }

  // ---- combine directions (unflip bwd), then output gating
  combine_mamba_kernel<<<(TTOK * Hdim + 255) / 256, 256, 0, stream>>>(
      gw, mf, mb, mcomb, mcombbf);

  // g = sigmoid([m_comb, f_out] @ gate_w^T + gate_b) as two accumulated GEMMs
  gemm(mcombbf, Hdim, w_gate,        2 * Hdim, gbuf, Hdim, nullptr, gate_b, Hdim, Hdim, 0, 0, 0);
  gemm(foutbf,  Hdim, w_gate + Hdim, 2 * Hdim, gbuf, Hdim, nullptr, nullptr, Hdim, Hdim, 0, 1, 1);

  combine2_kernel<<<(TTOK * Hdim + 255) / 256, 256, 0, stream>>>(
      gbuf, mcomb, fout, zsum, zsumbf, TTOK * Hdim);

  gemm(zsumbf, Hdim, w_mix, Hdim, z2, Hdim, nullptr, mix_b, Hdim, Hdim, 0, 0, 0);

  layernorm_res_kernel<<<TTOK / 8, 256, 0, stream>>>(
      z2, x, ln_g, ln_b, (float*)d_out, nullptr);
}
